// DynamicMultiheadAttention_54709293416664
// MI455X (gfx1250) — matrix operand order repaired and checked by reading, not compiled, not run on hardware
//
#include <hip/hip_runtime.h>
#include <hip/hip_bf16.h>
#include <math.h>

// ---------------------------------------------------------------------------
// CDNA5 (gfx1250) wave32 WMMA types
// ---------------------------------------------------------------------------
typedef __attribute__((ext_vector_type(16))) __bf16 v16bf;
typedef __attribute__((ext_vector_type(8)))  __bf16 v8bf;
typedef __attribute__((ext_vector_type(8)))  float  v8f;
typedef int v4i __attribute__((vector_size(16)));
typedef __attribute__((address_space(1))) v4i* as1_v4i;
typedef __attribute__((address_space(3))) v4i* as3_v4i;

__device__ __forceinline__ __bf16 f2bf(float f) {
    union { float f; unsigned u; } x; x.f = f;
    unsigned r = x.u + 0x7FFFu + ((x.u >> 16) & 1u);   // round-to-nearest-even
    unsigned short h = (unsigned short)(r >> 16);
    __bf16 out;
    __builtin_memcpy(&out, &h, 2);
    return out;
}

// A-fragment (16-bit 16x32, ISA 7.12.2): lane's 16 elems = two contiguous
// 8-element runs at k = 8g and k = 16 + 8g.
template <typename PT>
__device__ __forceinline__ v16bf load_a_frag(const PT* rowk0, int g) {
    v8bf lo = *(const v8bf*)(rowk0 + 8 * g);
    v8bf hi = *(const v8bf*)(rowk0 + 16 + 8 * g);
    return __builtin_shufflevector(lo, hi, 0, 1, 2, 3, 4, 5, 6, 7,
                                   8, 9, 10, 11, 12, 13, 14, 15);
}
// B-fragment (16-bit 32x16): lane's 16 elems = two contiguous
// 8-element runs at k = 8g and k = 16 + 8g.
template <typename PT>
__device__ __forceinline__ v16bf load_b_frag(const PT* colk0, int g) {
    v8bf lo = *(const v8bf*)(colk0 + 8 * g);
    v8bf hi = *(const v8bf*)(colk0 + 16 + 8 * g);
    return __builtin_shufflevector(lo, hi, 0, 1, 2, 3, 4, 5, 6, 7,
                                   8, 9, 10, 11, 12, 13, 14, 15);
}

// ---------------------------------------------------------------------------
// Async global -> LDS copy (GLOBAL_LOAD_ASYNC_TO_LDS_B128, ASYNCcnt-tracked)
// ---------------------------------------------------------------------------
__device__ __forceinline__ void async_b128(const void* gsrc, void* ldst) {
#if __has_builtin(__builtin_amdgcn_global_load_async_to_lds_b128)
    __builtin_amdgcn_global_load_async_to_lds_b128(
        (as1_v4i)(uintptr_t)gsrc,
        (as3_v4i)(unsigned)(uintptr_t)ldst,
        0, 0);
#else
    const unsigned laddr = (unsigned)(uintptr_t)ldst;       // LDS byte offset
    const unsigned long long gaddr = (unsigned long long)(uintptr_t)gsrc;
    asm volatile("global_load_async_to_lds_b128 %0, %1, off"
                 :: "v"(laddr), "v"(gaddr) : "memory");
#endif
}
__device__ __forceinline__ void wait_async_le2() {
#if __has_builtin(__builtin_amdgcn_s_wait_asynccnt)
    __builtin_amdgcn_s_wait_asynccnt(2);
#else
    asm volatile("s_wait_asynccnt 0x2" ::: "memory");
#endif
}
__device__ __forceinline__ void wait_async_0() {
#if __has_builtin(__builtin_amdgcn_s_wait_asynccnt)
    __builtin_amdgcn_s_wait_asynccnt(0);
#else
    asm volatile("s_wait_asynccnt 0x0" ::: "memory");
#endif
}

// ---------------------------------------------------------------------------
// f32 -> bf16 elementwise convert (vectorized by 4)
// ---------------------------------------------------------------------------
__global__ void __launch_bounds__(256)
cvt_bf16_kernel(const float* __restrict__ in, __bf16* __restrict__ out,
                size_t n4) {
    size_t i = (size_t)blockIdx.x * blockDim.x + threadIdx.x;
    if (i >= n4) return;
    const float4 v = ((const float4*)in)[i];
    __bf16* o = out + 4 * i;
    o[0] = f2bf(v.x); o[1] = f2bf(v.y); o[2] = f2bf(v.z); o[3] = f2bf(v.w);
}

// ---------------------------------------------------------------------------
// W (K x N, f32) -> Wt (N x K, bf16), LDS-tiled transpose
// ---------------------------------------------------------------------------
__global__ void __launch_bounds__(256)
transpose_w_kernel(const float* __restrict__ W, __bf16* __restrict__ Wt,
                   int K, int N) {
    __shared__ float tile[32][33];
    const int tx = threadIdx.x, ty = threadIdx.y;   // block (32, 8)
    const int n0 = blockIdx.x * 32, k0 = blockIdx.y * 32;
#pragma unroll
    for (int i = ty; i < 32; i += 8)
        tile[i][tx] = W[(size_t)(k0 + i) * N + n0 + tx];
    __syncthreads();
#pragma unroll
    for (int i = ty; i < 32; i += 8)
        Wt[(size_t)(n0 + i) * K + k0 + tx] = f2bf(tile[tx][i]);
}

// ---------------------------------------------------------------------------
// GEMM: out[M x Ncol] = A[M x K](bf16) @ Wt[Ncol x K](bf16, pre-transposed)
//       + bias
// mode 0: f32 row-major to outF
// mode 1: bf16 (b, h, n, c) headed layout to outH          (row m = n*Bdim+b)
// mode 2: bf16 (b, h, c, n) headed-TRANSPOSED layout (for V)
// Block = 4 waves = 64 rows x 64 cols. Shared 64x32 W tile is async-staged
// into double-buffered LDS (global_load_async_to_lds_b128 + s_wait_asynccnt),
// copy of tile k+1 overlapped with WMMAs on tile k.
// ---------------------------------------------------------------------------
__global__ void __launch_bounds__(128)
gemm_wmma(const __bf16* __restrict__ A, const __bf16* __restrict__ Wt,
          const float* __restrict__ bias,
          float* __restrict__ outF, __bf16* __restrict__ outH,
          int M, int K, int Ncol, int mode, int Bdim, int headC) {
    const int tid  = threadIdx.x;
    const int lane = tid & 31, wave = tid >> 5;
    const int m0 = (blockIdx.x * 4 + wave) * 16;
    const int n0 = blockIdx.y * 64;
    const int l15 = lane & 15, g = lane >> 4;

    __shared__ __align__(16) __bf16 bt[2][64][32];   // [buf][col][k] 4KB each

    const __bf16* arow = A + (size_t)(m0 + l15) * K;

    // cooperative fill: 128 threads x 2 x b128 = 4KB tile
    const int idx0 = tid * 2, idx1 = idx0 + 1;
    const int r0 = idx0 >> 2, c0 = idx0 & 3;         // row 0..63, 16B chunk 0..3
    const int r1 = idx1 >> 2, c1 = idx1 & 3;
    const __bf16* wbase = Wt + (size_t)n0 * K;

    // prologue: fill buffer 0 with k0 = 0
    async_b128(wbase + (size_t)r0 * K + c0 * 8, &bt[0][r0][c0 * 8]);
    async_b128(wbase + (size_t)r1 * K + c1 * 8, &bt[0][r1][c1 * 8]);

    v8f acc[4] = {};
    for (int k0 = 0; k0 < K; k0 += 32) {
        const int cur = (k0 >> 5) & 1, nxt = cur ^ 1;
        const v16bf a = load_a_frag(arow + k0, g);
        __syncthreads();                 // readers of bt[nxt] (iter-1) done
        if (k0 + 32 < K) {
            const __bf16* wb = wbase + (k0 + 32);
            async_b128(wb + (size_t)r0 * K + c0 * 8, &bt[nxt][r0][c0 * 8]);
            async_b128(wb + (size_t)r1 * K + c1 * 8, &bt[nxt][r1][c1 * 8]);
            wait_async_le2();            // cur fills done; nxt stays in flight
        } else {
            wait_async_0();
        }
        __syncthreads();                 // all waves' fills of bt[cur] visible
        // hoist all B-fragment DS loads, then issue WMMAs back-to-back
        v16bf bf[4];
#pragma unroll
        for (int t = 0; t < 4; ++t)
            bf[t] = load_b_frag(&bt[cur][t * 16 + l15][0], g);
#pragma unroll
        for (int t = 0; t < 4; ++t)
            acc[t] = __builtin_amdgcn_wmma_f32_16x16x32_bf16(
                false, a, false, bf[t], (short)0, acc[t], false, false);
    }

    // C layout: VGPR r, lane -> row m0 + r + 8g, col n0 + 16t + l15
#pragma unroll
    for (int t = 0; t < 4; ++t) {
        const int col = n0 + t * 16 + l15;
        const float bv = bias[col];
#pragma unroll
        for (int r = 0; r < 8; ++r) {
            const int m = m0 + r + 8 * g;
            const float val = acc[t][r] + bv;
            if (mode == 0) {
                outF[(size_t)m * Ncol + col] = val;
            } else {
                const int n = m / Bdim, b = m % Bdim;
                const int h = col / headC, c = col % headC;
                const int Nq = M / Bdim, H = Ncol / headC;
                if (mode == 1)
                    outH[(((size_t)b * H + h) * Nq + n) * headC + c] = f2bf(val);
                else  // mode 2: transposed (b,h,c,n)
                    outH[(((size_t)b * H + h) * headC + c) * (size_t)Nq + n] = f2bf(val);
            }
        }
    }
}

// ---------------------------------------------------------------------------
// Fused flash attention with relative bias.
// One wave per (b*H + h, 16-query-row tile).
// Q,K: bf16 (b,h,n,64). V: bf16 (b,h,64,n) TRANSPOSED. Out: bf16 (n,b,512).
// ---------------------------------------------------------------------------
__global__ void __launch_bounds__(32)
attn_kernel(const __bf16* __restrict__ Q, const __bf16* __restrict__ Kh,
            const __bf16* __restrict__ Vt,
            const unsigned char* __restrict__ amask,   // (B,R,N,N) bool
            const unsigned char* __restrict__ kpad,    // (B,N) bool
            const float* __restrict__ mw,              // (H, R+1)
            __bf16* __restrict__ Obf,                  // (N, B, H*64)
            int Nq, int Bb, int Hh) {
    const int lane = threadIdx.x & 31;
    const int l15 = lane & 15, g = lane >> 4;
    const int bh = blockIdx.x;
    const int b = bh / Hh, h = bh % Hh;
    const int n0 = blockIdx.y * 16;

    const __bf16* qp = Q + (size_t)bh * Nq * 64;
    const __bf16* kp = Kh + (size_t)bh * Nq * 64;
    const __bf16* vp = Vt + (size_t)bh * Nq * 64;   // (c, n) layout

    // per-head relative-bias weights: softmax(mw[h,0:3]) * mw[h,3]
    float w0 = mw[h * 4 + 0], w1 = mw[h * 4 + 1], w2 = mw[h * 4 + 2];
    float wm = fmaxf(w0, fmaxf(w1, w2));
    w0 = __expf(w0 - wm); w1 = __expf(w1 - wm); w2 = __expf(w2 - wm);
    const float ws = mw[h * 4 + 3] / (w0 + w1 + w2);
    const float rw[3] = { w0 * ws, w1 * ws, w2 * ws };

    // Q A-fragments (16 rows x 64 = 2 fragments of 16x32)
    v16bf qa[2];
#pragma unroll
    for (int kb = 0; kb < 2; ++kb)
        qa[kb] = load_a_frag(qp + (size_t)(n0 + l15) * 64 + 32 * kb, g);

    v8f o[4] = {};
    float rmax[8], rsum[8];
#pragma unroll
    for (int r = 0; r < 8; ++r) { rmax[r] = -INFINITY; rsum[r] = 0.f; }

    __shared__ __align__(16) __bf16 lp[16][48];   // P tile; 96B row stride

    for (int ks = 0; ks < Nq; ks += 32) {
        if (ks + 32 < Nq) {  // stream next K/V tiles (global_prefetch_b8)
            __builtin_prefetch(kp + (size_t)(ks + 32 + lane) * 64, 0, 1);
            __builtin_prefetch(vp + (size_t)(lane * 2) * Nq + ks + 32, 0, 1);
        }
        // ---- scores: two 16x16 tiles; all K-fragments loaded first --------
        v16bf kf[2][2];
#pragma unroll
        for (int t2 = 0; t2 < 2; ++t2) {
            const __bf16* krow = kp + (size_t)(ks + t2 * 16 + l15) * 64;
#pragma unroll
            for (int kb = 0; kb < 2; ++kb)
                kf[t2][kb] = load_b_frag(krow + 32 * kb, g);
        }
        v8f s[2] = {};
#pragma unroll
        for (int t2 = 0; t2 < 2; ++t2)
#pragma unroll
            for (int kb = 0; kb < 2; ++kb)
                s[t2] = __builtin_amdgcn_wmma_f32_16x16x32_bf16(
                    false, qa[kb], false, kf[t2][kb], (short)0, s[t2],
                    false, false);
        // ---- relative bias + key padding ----------------------------------
#pragma unroll
        for (int t2 = 0; t2 < 2; ++t2) {
            const int mcol = ks + t2 * 16 + l15;
            const bool pad = kpad[(size_t)b * Nq + mcol] != 0;
#pragma unroll
            for (int r = 0; r < 8; ++r) {
                float sv = s[t2][r] * 0.125f;   // 1/sqrt(64)
                if (pad) {
                    sv = -INFINITY;
                } else {
                    const int nrow = n0 + r + 8 * g;
                    const size_t mb = (((size_t)b * 3) * Nq + nrow) * Nq + mcol;
                    const size_t rs = (size_t)Nq * Nq;
                    sv += rw[0] * (amask[mb]          ? 0.f : 1.f);
                    sv += rw[1] * (amask[mb + rs]     ? 0.f : 1.f);
                    sv += rw[2] * (amask[mb + 2 * rs] ? 0.f : 1.f);
                }
                s[t2][r] = sv;
            }
        }
        // ---- online softmax (row = 16 lanes of one half) ------------------
        float p0[8], p1[8];
#pragma unroll
        for (int r = 0; r < 8; ++r) {
            float loc = fmaxf(s[0][r], s[1][r]);
#pragma unroll
            for (int off = 8; off > 0; off >>= 1)
                loc = fmaxf(loc, __shfl_xor(loc, off, 32));
            const float nm = fmaxf(rmax[r], loc);
            const float alpha = __expf(rmax[r] - nm);
            const float e0 = __expf(s[0][r] - nm);
            const float e1 = __expf(s[1][r] - nm);
            float ls = e0 + e1;
#pragma unroll
            for (int off = 8; off > 0; off >>= 1)
                ls += __shfl_xor(ls, off, 32);
            rsum[r] = rsum[r] * alpha + ls;
            rmax[r] = nm;
            p0[r] = e0; p1[r] = e1;
#pragma unroll
            for (int t = 0; t < 4; ++t) o[t][r] *= alpha;
        }
        // ---- repack P (C layout) -> A fragment via LDS --------------------
#pragma unroll
        for (int r = 0; r < 8; ++r) {
            lp[r + 8 * g][l15]      = f2bf(p0[r]);
            lp[r + 8 * g][16 + l15] = f2bf(p1[r]);
        }
        __syncthreads();
        const v16bf pa = load_a_frag(&lp[l15][0], g);
        __syncthreads();
        // ---- P @ V: all V-fragments loaded first --------------------------
        v16bf vb[4];
#pragma unroll
        for (int t = 0; t < 4; ++t)
            vb[t] = load_b_frag(vp + (size_t)(16 * t + l15) * Nq + ks, g);
#pragma unroll
        for (int t = 0; t < 4; ++t)
            o[t] = __builtin_amdgcn_wmma_f32_16x16x32_bf16(
                false, pa, false, vb[t], (short)0, o[t], false, false);
    }

    // ---- normalize + store (n, b, h*64 + c) bf16 --------------------------
#pragma unroll
    for (int t = 0; t < 4; ++t) {
        const int col = h * 64 + 16 * t + l15;
#pragma unroll
        for (int r = 0; r < 8; ++r) {
            const int nrow = n0 + r + 8 * g;
            const float inv = (rsum[r] > 0.f) ? (1.f / rsum[r]) : 0.f;
            Obf[((size_t)nrow * Bb + b) * (size_t)(Hh * 64) + col] =
                f2bf(o[t][r] * inv);
        }
    }
}

// ---------------------------------------------------------------------------
extern "C" void kernel_launch(void* const* d_in, const int* in_sizes, int n_in,
                              void* d_out, int out_size, void* d_ws, size_t ws_size,
                              hipStream_t stream) {
    constexpr int N = 2048, B = 2, D = 512, H = 8, C = 64;
    const int M = N * B;

    const float*         x_q  = (const float*)d_in[0];
    const float*         x_k  = (const float*)d_in[1];
    const float*         x_v  = (const float*)d_in[2];
    const unsigned char* am   = (const unsigned char*)d_in[3];
    const unsigned char* kpad = (const unsigned char*)d_in[4];
    const float* Wq = (const float*)d_in[5];
    const float* bq = (const float*)d_in[6];
    const float* Wk = (const float*)d_in[7];
    const float* bk = (const float*)d_in[8];
    const float* Wv = (const float*)d_in[9];
    const float* bv = (const float*)d_in[10];
    const float* Wo = (const float*)d_in[11];
    const float* bo = (const float*)d_in[12];
    const float* mw = (const float*)d_in[13];

    // Workspace layout (bf16 elements):
    const size_t tokE = (size_t)M * D;        // 2,097,152 (x / q / k / v / o)
    const size_t wE   = (size_t)D * D;        //   262,144 per weight
    __bf16* p    = (__bf16*)d_ws;
    __bf16* xqb  = p;  p += tokE;
    __bf16* xkb  = p;  p += tokE;
    __bf16* xvb  = p;  p += tokE;
    __bf16* qbf  = p;  p += tokE;
    __bf16* kbf  = p;  p += tokE;
    __bf16* vbf  = p;  p += tokE;   // transposed (b,h,c,n)
    __bf16* obf  = p;  p += tokE;
    __bf16* wqT  = p;  p += wE;
    __bf16* wkT  = p;  p += wE;
    __bf16* wvT  = p;  p += wE;
    __bf16* woT  = p;  p += wE;

    // 1) convert activations to bf16
    {
        const size_t n4 = tokE / 4;
        const int blk = 256, grd = (int)((n4 + blk - 1) / blk);
        cvt_bf16_kernel<<<grd, blk, 0, stream>>>(x_q, xqb, n4);
        cvt_bf16_kernel<<<grd, blk, 0, stream>>>(x_k, xkb, n4);
        cvt_bf16_kernel<<<grd, blk, 0, stream>>>(x_v, xvb, n4);
    }
    // 2) transpose + convert weights
    {
        dim3 tb(32, 8), tg(D / 32, D / 32);
        transpose_w_kernel<<<tg, tb, 0, stream>>>(Wq, wqT, D, D);
        transpose_w_kernel<<<tg, tb, 0, stream>>>(Wk, wkT, D, D);
        transpose_w_kernel<<<tg, tb, 0, stream>>>(Wv, wvT, D, D);
        transpose_w_kernel<<<tg, tb, 0, stream>>>(Wo, woT, D, D);
    }
    // 3) QKV projections (WMMA + async LDS staging)
    dim3 gblk(M / 64, D / 64);   // 64 x 8 blocks, 128 threads (4 waves)
    gemm_wmma<<<gblk, 128, 0, stream>>>(xqb, wqT, bq, nullptr, qbf,
                                        M, D, D, 1, B, C);
    gemm_wmma<<<gblk, 128, 0, stream>>>(xkb, wkT, bk, nullptr, kbf,
                                        M, D, D, 1, B, C);
    gemm_wmma<<<gblk, 128, 0, stream>>>(xvb, wvT, bv, nullptr, vbf,
                                        M, D, D, 2, B, C);
    // 4) fused flash attention (WMMA)
    attn_kernel<<<dim3(B * H, N / 16), 32, 0, stream>>>(
        qbf, kbf, vbf, am, kpad, mw, obf, N, B, H);
    // 5) output projection (WMMA)
    gemm_wmma<<<gblk, 128, 0, stream>>>(obf, woT, bo, (float*)d_out,
                                        nullptr, M, D, D, 0, B, C);
}
